// MultiHeadAttention_89627377533577
// MI455X (gfx1250) — compile-verified, matrix operand order checked by reading, not run on hardware
//
#include <hip/hip_runtime.h>
#include <cstdint>
#include <cstddef>

typedef unsigned short u16;
typedef unsigned int   u32;
typedef __attribute__((ext_vector_type(16))) __bf16 v16bf;
typedef __attribute__((ext_vector_type(8)))  float  v8f;
typedef __attribute__((ext_vector_type(4)))  unsigned int v4u;
typedef __attribute__((ext_vector_type(8)))  int v8i_t;
typedef __attribute__((ext_vector_type(4)))  int v4i_t;

#define NB_B  2
#define NB_T  2048
#define NB_D  2048
#define NB_H  16
#define NB_KV 4
#define NB_DH 128

// ---------- helpers ----------

__device__ __forceinline__ u16 f2bf(float f) {          // fp32 -> bf16 RNE
  unsigned u = __builtin_bit_cast(unsigned, f);
  unsigned r = ((u >> 16) & 1u) + 0x7FFFu;
  return (u16)((u + r) >> 16);
}

union FragU { v16bf v; uint4 u[2]; };

// Load a 16x32 bf16 fragment in the CDNA5 WMMA A/B register layout.
// Works for A (rows = M) and for B when the matrix is stored (N,K) row-major
// (i.e. B column-major), which is the case for all our operands.
// Lane<16: K = kBase+[0..7] and kBase+[16..23]; lane>=16: K = +8 / +24.
__device__ __forceinline__ v16bf load_tile16x32(const u16* base, int ld,
                                                int rowBase, int kBase, int lane) {
  const int hl = lane >> 4;
  const int rr = lane & 15;
  const u16* p = base + (size_t)(rowBase + rr) * (size_t)ld + (size_t)(kBase + hl * 8);
  FragU f;
  f.u[0] = *(const uint4*)(p);
  f.u[1] = *(const uint4*)(p + 16);
  return f.v;
}

__device__ __forceinline__ float hrmax16(float v) {     // max over 16-lane half
  v = fmaxf(v, __shfl_xor(v, 1, 32));
  v = fmaxf(v, __shfl_xor(v, 2, 32));
  v = fmaxf(v, __shfl_xor(v, 4, 32));
  v = fmaxf(v, __shfl_xor(v, 8, 32));
  return v;
}
__device__ __forceinline__ float hradd16(float v) {     // sum over 16-lane half
  v += __shfl_xor(v, 1, 32);
  v += __shfl_xor(v, 2, 32);
  v += __shfl_xor(v, 4, 32);
  v += __shfl_xor(v, 8, 32);
  return v;
}

// ---------- Tensor Data Mover: 2D tile global -> LDS ----------
// Builds D# group0/group1 per CDNA5 ISA (data_size = 2B,
// count=1, type=2 "image", no gather/iterate/pad, workgroup_mask=0).
// tdim0/tile0 = row length (elements), tdim1/tile1 = #rows,
// stride0 = elements between consecutive rows of the tensor.
__device__ __forceinline__ void tdm_load_2d(u32 lds_addr, const void* gptr,
                                            u32 tdim0, u32 tdim1,
                                            u32 tile0, u32 tile1, u32 stride0) {
  unsigned long long ga = (unsigned long long)(uintptr_t)gptr;
  v4u g0;
  g0[0] = 1u;                                          // count=1 (valid), user mode
  g0[1] = lds_addr;                                    // LDS byte address
  g0[2] = (u32)ga;                                     // global_addr[31:0]
  g0[3] = ((u32)(ga >> 32) & 0x01FFFFFFu) | (2u << 30);// global_addr[56:32] | type=2
  v8i_t g1;
  g1[0] = (int)(1u << 16);                             // data_size=1 -> 2 bytes
  g1[1] = (int)((tdim0 & 0xFFFFu) << 16);              // tensor_dim0[15:0]
  g1[2] = (int)(((tdim0 >> 16) & 0xFFFFu) | ((tdim1 & 0xFFFFu) << 16));
  g1[3] = (int)(((tdim1 >> 16) & 0xFFFFu) | ((tile0 & 0xFFFFu) << 16));
  g1[4] = (int)(tile1 & 0xFFFFu);                      // tile_dim1 (tile_dim2=0)
  g1[5] = (int)stride0;                                // tensor_dim0_stride[31:0]
  g1[6] = 0;
  g1[7] = 0;
  v4i_t g2 = {0, 0, 0, 0};
  v4i_t g3 = {0, 0, 0, 0};
#if defined(__clang_major__) && (__clang_major__ >= 23)
  v8i_t g4 = {0, 0, 0, 0, 0, 0, 0, 0};
  __builtin_amdgcn_tensor_load_to_lds(g0, g1, g2, g3, g4, 0);
#else
  __builtin_amdgcn_tensor_load_to_lds(g0, g1, g2, g3, 0);
#endif
}

// ---------- fp32 -> bf16 convert ----------

__global__ __launch_bounds__(256) void cvt_bf16_kernel(const float* __restrict__ in,
                                                       u16* __restrict__ out, size_t n) {
  size_t i = (size_t)blockIdx.x * 256 + threadIdx.x;
  if (i < n) out[i] = f2bf(in[i]);
}

// ---------- generic bf16 WMMA GEMM: C[M,N] = A[M,K] * W[N,K]^T ----------
// block = 256 threads (8 waves), tile 128(M) x 128(N), k-panel 32.
// TDM double-buffered LDS staging: wave 0 issues DMA for panel kt+1 while all
// waves run WMMA on panel kt; s_wait_tensorcnt + barrier synchronize.

__global__ __launch_bounds__(256) void gemm_bf16_nt(const u16* __restrict__ A,
                                                    const u16* __restrict__ W,
                                                    float* __restrict__ C,
                                                    int M, int N, int K) {
  __shared__ alignas(16) u16 As[2][128 * 32];
  __shared__ alignas(16) u16 Ws[2][128 * 32];
  const int m0 = blockIdx.y * 128, n0 = blockIdx.x * 128;
  const int tid = threadIdx.x, wave = tid >> 5, lane = tid & 31;
  const int wm = wave >> 1, wn = wave & 1;
  const int hl = lane >> 4, ln = lane & 15;
  (void)M;

  const u32 ldsA0 = (u32)(uintptr_t)&As[0][0], ldsA1 = (u32)(uintptr_t)&As[1][0];
  const u32 ldsW0 = (u32)(uintptr_t)&Ws[0][0], ldsW1 = (u32)(uintptr_t)&Ws[1][0];

  v8f acc[2][4] = {};
  const int nk = K >> 5;

  if (wave == 0) {
    tdm_load_2d(ldsA0, A + (size_t)m0 * K, 32, 128, 32, 128, (u32)K);
    tdm_load_2d(ldsW0, W + (size_t)n0 * K, 32, 128, 32, 128, (u32)K);
  }

  for (int kt = 0; kt < nk; ++kt) {
    const int cur = kt & 1;
    if (wave == 0) {
      if (kt + 1 < nk) {
        const int k1 = (kt + 1) << 5;
        tdm_load_2d(cur ? ldsA0 : ldsA1, A + (size_t)m0 * K + k1, 32, 128, 32, 128, (u32)K);
        tdm_load_2d(cur ? ldsW0 : ldsW1, W + (size_t)n0 * K + k1, 32, 128, 32, 128, (u32)K);
        __builtin_amdgcn_s_wait_tensorcnt((short)2);   // panel kt complete (in-order)
      } else {
        __builtin_amdgcn_s_wait_tensorcnt((short)0);
      }
    }
    __syncthreads();

    v16bf af[2], wf[4];
#pragma unroll
    for (int i = 0; i < 2; i++) af[i] = load_tile16x32(&As[cur][0], 32, wm * 32 + i * 16, 0, lane);
#pragma unroll
    for (int j = 0; j < 4; j++) wf[j] = load_tile16x32(&Ws[cur][0], 32, wn * 64 + j * 16, 0, lane);

#pragma unroll
    for (int i = 0; i < 2; i++)
#pragma unroll
      for (int j = 0; j < 4; j++)
        acc[i][j] = __builtin_amdgcn_wmma_f32_16x16x32_bf16(
            false, af[i], false, wf[j], (short)0, acc[i][j], false, false);
    __syncthreads();
  }

#pragma unroll
  for (int i = 0; i < 2; i++)
#pragma unroll
    for (int j = 0; j < 4; j++) {
      int mb = m0 + wm * 32 + i * 16 + hl * 8;
      int nn = n0 + wn * 64 + j * 16 + ln;
#pragma unroll
      for (int e = 0; e < 8; e++)
        C[(size_t)(mb + e) * N + nn] = acc[i][j][e];
    }
}

// ---------- RoPE (+ optional score-scale fold) ----------
// X: (B, T, nH, DH) fp32  ->  Y: (B, nH, T, DH) bf16

__global__ __launch_bounds__(256) void rope_kernel(const float* __restrict__ X,
                                                   u16* __restrict__ Y,
                                                   int nH, float scale, size_t n) {
  size_t idx = (size_t)blockIdx.x * 256 + threadIdx.x;
  if (idx >= n) return;
  int d = (int)(idx & 127);
  size_t r = idx >> 7;
  int hh = (int)(r % nH);
  size_t r2 = r / nH;
  int t = (int)(r2 % NB_T);
  int b = (int)(r2 / NB_T);
  int j = d & 63;
  bool hi = d >= 64;
  float invf = __powf(10000.0f, -(float)j * (1.0f / 64.0f));
  float ang = (float)t * invf;
  float s, c;
  __sincosf(ang, &s, &c);
  float x  = X[idx];
  float xo = X[idx + (hi ? -64 : 64)];
  float val = hi ? (x * c + xo * s) : (x * c - xo * s);
  Y[((((size_t)b * nH + hh) * NB_T) + t) * NB_DH + d] = f2bf(val * scale);
}

// ---------- V transpose: (B,T,KV,DH) fp32 -> (B,KV,DH,T) bf16 ----------

__global__ __launch_bounds__(256) void vtrans_kernel(const float* __restrict__ Vf,
                                                     u16* __restrict__ Vt, size_t n) {
  size_t idx = (size_t)blockIdx.x * 256 + threadIdx.x;  // output-linear
  if (idx >= n) return;
  int t = (int)(idx % NB_T);
  size_t r = idx / NB_T;
  int d  = (int)(r % NB_DH); r /= NB_DH;
  int kv = (int)(r % NB_KV);
  int b  = (int)(r / NB_KV);
  Vt[idx] = f2bf(Vf[((((size_t)b * NB_T + t) * NB_KV) + kv) * NB_DH + d]);
}

// ---------- flash attention (causal, GQA) ----------
// grid (T/128, H, B), block 256 (8 waves). Each wave owns a 16-row q tile.
// K/V^T tiles arrive via TDM double buffering (wave 0 issues, tensorcnt sync).
// Q: (B,H,T,DH) bf16 (pre-scaled by 1/sqrt(DH)), K: (B,KV,T,DH) bf16,
// Vt: (B,KV,DH,T) bf16. O: (B,T,H*DH) bf16.

__global__ __launch_bounds__(256) void mha_flash_kernel(const u16* __restrict__ Q,
                                                        const u16* __restrict__ Kc,
                                                        const u16* __restrict__ Vt,
                                                        u16* __restrict__ O) {
  __shared__ alignas(16) u16 Ks[2][32 * NB_DH];    // 32 keys x 128 dh
  __shared__ alignas(16) u16 Vs[2][NB_DH * 32];    // 128 dh x 32 keys (V^T tile)
  __shared__ alignas(16) u16 Ps[8 * 16 * 32];      // per-wave P tile
  const int b = blockIdx.z;
  const int h = blockIdx.y;
  const int q0 = blockIdx.x * 128;
  const int kvh = h >> 2;                          // REP = H/KV = 4
  const int tid = threadIdx.x, wave = tid >> 5, lane = tid & 31;
  const int hl = lane >> 4, ln = lane & 15;

  const u16* qptr = Q  + (((size_t)b * NB_H  + h)   * NB_T)  * NB_DH;
  const u16* kptr = Kc + (((size_t)b * NB_KV + kvh) * NB_T)  * NB_DH;
  const u16* vptr = Vt + (((size_t)b * NB_KV + kvh) * NB_DH) * NB_T;

  const u32 ldsK0 = (u32)(uintptr_t)&Ks[0][0], ldsK1 = (u32)(uintptr_t)&Ks[1][0];
  const u32 ldsV0 = (u32)(uintptr_t)&Vs[0][0], ldsV1 = (u32)(uintptr_t)&Vs[1][0];

  v16bf qf[4];
#pragma unroll
  for (int kk = 0; kk < 4; kk++)
    qf[kk] = load_tile16x32(qptr, NB_DH, q0 + wave * 16, kk * 32, lane);

  v8f oacc[8] = {};
  float mrow[8], lrow[8];
#pragma unroll
  for (int i = 0; i < 8; i++) { mrow[i] = -1e30f; lrow[i] = 0.0f; }

  const int qrBase = q0 + wave * 16 + hl * 8;
  const int nIter = (q0 + 128) >> 5;               // key tiles of 32

  if (wave == 0) {                                 // prologue: tile 0 DMA
    tdm_load_2d(ldsK0, kptr, 128, 32, 128, 32, 128);
    tdm_load_2d(ldsV0, vptr, 32, 128, 32, 128, NB_T);
  }

  for (int it = 0; it < nIter; ++it) {
    const int s0 = it << 5;
    const int cur = it & 1;
    if (wave == 0) {
      if (it + 1 < nIter) {                        // issue next tile into other buffer
        const int s1 = s0 + 32;
        tdm_load_2d(cur ? ldsK0 : ldsK1, kptr + (size_t)s1 * NB_DH, 128, 32, 128, 32, 128);
        tdm_load_2d(cur ? ldsV0 : ldsV1, vptr + s1, 32, 128, 32, 128, NB_T);
        __builtin_amdgcn_s_wait_tensorcnt((short)2);  // tile `it` complete (in-order)
      } else {
        __builtin_amdgcn_s_wait_tensorcnt((short)0);
      }
    }
    __syncthreads();

    // scores: 16 q-rows x 32 keys, two 16x16 WMMA tiles, K=128 in 4 steps
    v8f sc[2] = {};
#pragma unroll
    for (int j = 0; j < 2; j++)
#pragma unroll
      for (int kk = 0; kk < 4; kk++) {
        v16bf kf = load_tile16x32(&Ks[cur][0], NB_DH, j * 16, kk * 32, lane);
        sc[j] = __builtin_amdgcn_wmma_f32_16x16x32_bf16(
            false, qf[kk], false, kf, (short)0, sc[j], false, false);
      }

    // online softmax over the 8 rows this lane tracks (rows hl*8+i)
#pragma unroll
    for (int i = 0; i < 8; i++) {
      const int qr = qrBase + i;
#pragma unroll
      for (int j = 0; j < 2; j++) {
        int scol = s0 + j * 16 + ln;
        if (scol > qr) sc[j][i] = -1e30f;          // causal mask
      }
      float tmax = hrmax16(fmaxf(sc[0][i], sc[1][i]));
      float mnew = fmaxf(mrow[i], tmax);
      float corr = __expf(mrow[i] - mnew);
      float e0 = __expf(sc[0][i] - mnew);
      float e1 = __expf(sc[1][i] - mnew);
      sc[0][i] = e0;
      sc[1][i] = e1;
      lrow[i] = lrow[i] * corr + hradd16(e0 + e1);
      mrow[i] = mnew;
#pragma unroll
      for (int nt = 0; nt < 8; nt++) oacc[nt][i] = oacc[nt][i] * corr;
    }

    // P: C/D layout -> A layout via per-wave LDS round-trip
    u16* Pw = &Ps[wave * 512];
#pragma unroll
    for (int i = 0; i < 8; i++) {
      int rr = hl * 8 + i;
      Pw[rr * 32 + ln]      = f2bf(sc[0][i]);
      Pw[rr * 32 + 16 + ln] = f2bf(sc[1][i]);
    }
    v16bf pf = load_tile16x32(Pw, 32, 0, 0, lane);

    // O += P (16x32) x V (32x128): 8 WMMAs, single k-step
#pragma unroll
    for (int nt = 0; nt < 8; nt++) {
      v16bf vf = load_tile16x32(&Vs[cur][0], 32, nt * 16, 0, lane);
      oacc[nt] = __builtin_amdgcn_wmma_f32_16x16x32_bf16(
          false, pf, false, vf, (short)0, oacc[nt], false, false);
    }
    __syncthreads();
  }

  // normalize + store bf16 (B, T, H*DH)
#pragma unroll
  for (int i = 0; i < 8; i++) {
    float inv = 1.0f / lrow[i];
    int t = qrBase + i;
    u16* orow = O + ((size_t)b * NB_T + t) * NB_D + h * NB_DH;
#pragma unroll
    for (int nt = 0; nt < 8; nt++)
      orow[nt * 16 + ln] = f2bf(oacc[nt][i] * inv);
  }
}

// ---------- launcher ----------

extern "C" void kernel_launch(void* const* d_in, const int* in_sizes, int n_in,
                              void* d_out, int out_size, void* d_ws, size_t ws_size,
                              hipStream_t stream) {
  (void)in_sizes; (void)n_in; (void)out_size; (void)ws_size;
  const float* x  = (const float*)d_in[0];   // (2, 2048, 2048)
  const float* wq = (const float*)d_in[1];   // (2048, 2048)
  const float* wk = (const float*)d_in[2];   // (512, 2048)
  const float* wv = (const float*)d_in[3];   // (512, 2048)
  const float* wo = (const float*)d_in[4];   // (2048, 2048)
  float* out = (float*)d_out;                // (2, 2048, 2048)

  char* w = (char*)d_ws;
  u16*   xb  = (u16*)  (w + 0);              // x bf16            16 MB
  u16*   wqb = (u16*)  (w + 16777216);       // wq bf16            8 MB
  u16*   wkb = (u16*)  (w + 25165824);       // wk bf16            2 MB
  u16*   wvb = (u16*)  (w + 27262976);       // wv bf16            2 MB
  u16*   wob = (u16*)  (w + 29360128);       // wo bf16            8 MB
  float* Qf  = (float*)(w + 37748736);       // q fp32 pre-rope   32 MB
  float* Kf  = (float*)(w + 71303168);       // k fp32 pre-rope    8 MB
  float* Vf  = (float*)(w + 79691776);       // v fp32             8 MB
  u16*   qb  = (u16*)  (w + 88080384);       // q bf16 (B,H,T,DH) 16 MB
  u16*   kb  = (u16*)  (w + 104857600);      // k bf16 (B,KV,T,DH) 4 MB
  u16*   vtb = (u16*)  (w + 109051904);      // v^T bf16           4 MB
  u16*   Ab  = (u16*)  (w + 113246208);      // attn out bf16     16 MB

  const size_t nX = (size_t)NB_B * NB_T * NB_D;        // 8388608
  const size_t nWq = (size_t)NB_D * NB_D;              // 4194304
  const size_t nWk = (size_t)NB_KV * NB_DH * NB_D;     // 1048576

  cvt_bf16_kernel<<<(unsigned)((nX  + 255) / 256), 256, 0, stream>>>(x,  xb,  nX);
  cvt_bf16_kernel<<<(unsigned)((nWq + 255) / 256), 256, 0, stream>>>(wq, wqb, nWq);
  cvt_bf16_kernel<<<(unsigned)((nWk + 255) / 256), 256, 0, stream>>>(wk, wkb, nWk);
  cvt_bf16_kernel<<<(unsigned)((nWk + 255) / 256), 256, 0, stream>>>(wv, wvb, nWk);
  cvt_bf16_kernel<<<(unsigned)((nWq + 255) / 256), 256, 0, stream>>>(wo, wob, nWq);

  // projections (M = B*T = 4096)
  gemm_bf16_nt<<<dim3(NB_D / 128, 4096 / 128), 256, 0, stream>>>(xb, wqb, Qf, 4096, NB_D, NB_D);
  gemm_bf16_nt<<<dim3(512  / 128, 4096 / 128), 256, 0, stream>>>(xb, wkb, Kf, 4096, 512, NB_D);
  gemm_bf16_nt<<<dim3(512  / 128, 4096 / 128), 256, 0, stream>>>(xb, wvb, Vf, 4096, 512, NB_D);

  // rope (fold 1/sqrt(DH) into q) + layout changes
  const float qscale = 0.088388347648318447f;          // 1/sqrt(128)
  rope_kernel<<<(unsigned)(nX / 256), 256, 0, stream>>>(Qf, qb, NB_H, qscale, nX);
  const size_t nK = (size_t)NB_B * NB_KV * NB_T * NB_DH;   // 2097152
  rope_kernel<<<(unsigned)(nK / 256), 256, 0, stream>>>(Kf, kb, NB_KV, 1.0f, nK);
  vtrans_kernel<<<(unsigned)(nK / 256), 256, 0, stream>>>(Vf, vtb, nK);

  // attention
  mha_flash_kernel<<<dim3(NB_T / 128, NB_H, NB_B), 256, 0, stream>>>(qb, kb, vtb, Ab);

  // output projection -> fp32 d_out
  gemm_bf16_nt<<<dim3(NB_D / 128, 4096 / 128), 256, 0, stream>>>(Ab, wob, out, 4096, NB_D, NB_D);
}
